// DigitCaps_89300960018600
// MI455X (gfx1250) — matrix operand order repaired and checked by reading, not compiled, not run on hardware
//
#include <hip/hip_runtime.h>
#include <hip/hip_bf16.h>
#include <stdint.h>

// Problem constants (from reference): B=64, I=1024, K(A_in)=64, O=64, A=32, N=O*A=2048
#define BB 64
#define II 1024
#define KK 64
#define OO 64
#define AA 32
#define NN 2048

typedef __attribute__((ext_vector_type(16))) __bf16 v16bf;
typedef __attribute__((ext_vector_type(8)))  float  v8f;

__device__ __forceinline__ float wave_allsum(float v) {
#pragma unroll
  for (int off = 16; off > 0; off >>= 1) v += __shfl_xor(v, off, 32);
  return v;
}
__device__ __forceinline__ float wave_allmax(float v) {
#pragma unroll
  for (int off = 16; off > 0; off >>= 1) v = fmaxf(v, __shfl_xor(v, off, 32));
  return v;
}

// ---------------------------------------------------------------------------
// Kernel 1: votes[b,i,n] = sum_k inputs[b,i,k] * W[i,k,n], stored bf16.
// One block per i. 8 waves. Per 128-wide N step:
//   - stage W[k=0..63, 128 cols] via coalesced float4 loads -> bf16 in LDS
//   - prefetch next W chunk (global_prefetch_b8)
//   - each wave: gather B fragments from LDS, 4 M-tiles x 2 WMMA (K=64)
//   - stage D tiles in LDS, block-cooperative b128 stores (256B segments)
// ---------------------------------------------------------------------------
__global__ __launch_bounds__(256) void votes_gemm(const float* __restrict__ inp,
                                                  const float* __restrict__ W,
                                                  __bf16* __restrict__ votes) {
  __shared__ __bf16 As[64][66];                 // inputs[:, i, :] (bf16)
  __shared__ __bf16 Ws[64][132];                // staged W chunk (bf16)
  __shared__ __align__(16) __bf16 Vs[64][136];  // staged output slab (bf16)

  const int i    = blockIdx.x;
  const int tid  = threadIdx.x;
  const int wave = tid >> 5;
  const int lane = tid & 31;
  const int hh   = lane >> 4;    // wave half: K-group select
  const int l15  = lane & 15;

  for (int t = tid; t < 64 * 64; t += 256) {
    int b = t >> 6, k = t & 63;
    As[b][k] = (__bf16)inp[((size_t)b * II + i) * KK + k];
  }
  __syncthreads();

  // Preload all A fragments (16x32 MxK): row = l15; K map e + 8*hh + (e>=8 ? 8 : 0)
  v16bf a0[4], a1[4];
#pragma unroll
  for (int mt = 0; mt < 4; ++mt) {
    v16bf t0 = {}, t1 = {};
#pragma unroll
    for (int e = 0; e < 16; ++e) {
      const int ka = e + 8 * hh + ((e >= 8) ? 8 : 0);
      t0[e] = As[mt * 16 + l15][ka];
      t1[e] = As[mt * 16 + l15][ka + 32];
    }
    a0[mt] = t0;
    a1[mt] = t1;
  }

  const float* Wi = W + (size_t)i * KK * NN;

  for (int ns = 0; ns < 16; ++ns) {
    __syncthreads();  // Ws/Vs safe to overwrite
    // ---- stage W chunk: k=0..63, cols ns*128..+128 (coalesced float4) ----
    for (int t = tid; t < 64 * 32; t += 256) {
      const int k  = t >> 5;        // 0..63
      const int c4 = t & 31;        // 0..31 (4 floats)
      const float4 w4 = ((const float4*)(Wi + (size_t)k * NN + ns * 128))[c4];
      const int col = c4 * 4;
      Ws[k][col + 0] = (__bf16)w4.x;
      Ws[k][col + 1] = (__bf16)w4.y;
      Ws[k][col + 2] = (__bf16)w4.z;
      Ws[k][col + 3] = (__bf16)w4.w;
      if (ns + 1 < 16)
        __builtin_prefetch(Wi + (size_t)k * NN + (ns + 1) * 128 + col, 0, 1);
    }
    __syncthreads();

    // ---- compute: B fragments from LDS, 2 WMMA per M-tile ----
    const int n0 = wave * 16 + l15;
    v16bf b0 = {}, b1 = {};
#pragma unroll
    for (int e = 0; e < 16; ++e) {
      const int kb = e + 8 * hh + ((e >= 8) ? 8 : 0);   // B layout: same K map as A
      b0[e] = Ws[kb][n0];
      b1[e] = Ws[kb + 32][n0];
    }
#pragma unroll
    for (int mt = 0; mt < 4; ++mt) {
      v8f c = {};
      c = __builtin_amdgcn_wmma_f32_16x16x32_bf16(false, a0[mt], false, b0, (short)0, c, false, false);
      c = __builtin_amdgcn_wmma_f32_16x16x32_bf16(false, a1[mt], false, b1, (short)0, c, false, false);
      // D layout: VGPR r -> row r + 8*hh, col = l15
#pragma unroll
      for (int r = 0; r < 8; ++r)
        Vs[mt * 16 + r + 8 * hh][n0] = (__bf16)c[r];
    }
    __syncthreads();

    // ---- block-cooperative wide store: 64 rows x 128 cols bf16 ----
    for (int t = tid; t < 64 * 16; t += 256) {
      const int row = t >> 4;       // b index
      const int seg = t & 15;       // 8-element (16B) segment
      const uint4 v = *(const uint4*)&Vs[row][seg * 8];
      *(uint4*)(votes + ((size_t)row * II + i) * NN + ns * 128 + seg * 8) = v;
    }
  }
}

// ---------------------------------------------------------------------------
// Kernel 2: S[b,o,a] = sum_i votes; act0 = squash(S/65 + bias)   (iter 0)
// Thread <-> (b, o*32+a); a == lane, so one wave = one (b,o) atom vector.
// ---------------------------------------------------------------------------
__global__ __launch_bounds__(256) void reduce_act0(const __bf16* __restrict__ votes,
                                                   const float* __restrict__ biases,
                                                   float* __restrict__ act0) {
  const int idx = blockIdx.x * 256 + threadIdx.x;  // [0, B*N)
  const int b  = idx >> 11;
  const int oa = idx & (NN - 1);
  const __bf16* p = votes + (size_t)b * II * NN + oa;
  float s = 0.f;
  for (int i = 0; i < II; ++i) s += (float)p[(size_t)i * NN];
  const float x  = s * (1.0f / 65.0f) + biases[oa];
  const float ss = wave_allsum(x * x);
  act0[idx] = x * sqrtf(ss) / (1.0f + ss);
}

// ---------------------------------------------------------------------------
// Kernel 3 (x2): fused routing pass.
//   logits = (prev logits) + sum_a votes[b,i,o,a]*act[b,o,a]
//   route  = leaky_softmax(logits)
//   P[b,o,a] += sum_i route[b,i,o]*votes[b,i,o,a]
// Block = (b, 128-i chunk); wave processes one i at a time; lane == atom.
// ---------------------------------------------------------------------------
__global__ __launch_bounds__(256) void routing_pass(const __bf16* __restrict__ votes,
                                                    const float* __restrict__ act,
                                                    float* __restrict__ logits_store,
                                                    const float* __restrict__ logits_prev,
                                                    float* __restrict__ P) {
  __shared__ float  act_s[NN];
  __shared__ float  P_s[NN];
  __shared__ __bf16 vrow[8][OO][AA];
  __shared__ float  logit_s[8][OO];
  __shared__ float  route_s[8][OO];

  const int b      = blockIdx.x >> 3;
  const int ichunk = blockIdx.x & 7;
  const int tid    = threadIdx.x;
  const int wave   = tid >> 5;
  const int lane   = tid & 31;

  for (int t = tid; t < NN; t += 256) {
    act_s[t] = act[b * NN + t];
    P_s[t]   = 0.0f;
  }
  __syncthreads();

  for (int step = 0; step < 16; ++step) {
    const int i = ichunk * 128 + step * 8 + wave;
    // stage this wave's votes row [64][32] bf16 into LDS (coalesced u32 loads)
    const uint32_t* src = (const uint32_t*)(votes + ((size_t)b * II + i) * NN);
    uint32_t* dst = (uint32_t*)&vrow[wave][0][0];
#pragma unroll 4
    for (int t = lane; t < NN / 2; t += 32) dst[t] = src[t];
    __syncthreads();

    // agreement logits: wave reduction over atoms (lane == a)
    for (int o = 0; o < OO; ++o) {
      float x = (float)vrow[wave][o][lane] * act_s[o * AA + lane];
      x = wave_allsum(x);
      if (lane == 0) logit_s[wave][o] = x;
    }

    // leaky softmax over {leak=0, 64 logits}; lane owns o=lane and o=lane+32
    float l0 = logit_s[wave][lane];
    float l1 = logit_s[wave][lane + 32];
    if (logits_prev) {
      const float* lp = logits_prev + ((size_t)b * II + i) * OO;
      l0 += lp[lane];
      l1 += lp[lane + 32];
    }
    if (logits_store) {
      float* ls = logits_store + ((size_t)b * II + i) * OO;
      ls[lane]      = l0;
      ls[lane + 32] = l1;
    }
    float m = wave_allmax(fmaxf(fmaxf(l0, l1), 0.0f));
    const float e0 = __expf(l0 - m);
    const float e1 = __expf(l1 - m);
    const float Z  = wave_allsum(e0 + e1) + __expf(-m);
    route_s[wave][lane]      = e0 / Z;
    route_s[wave][lane + 32] = e1 / Z;

    // accumulate route * votes into block-local partial (ds_add_f32)
#pragma unroll 4
    for (int o = 0; o < OO; ++o) {
      atomicAdd(&P_s[o * AA + lane], route_s[wave][o] * (float)vrow[wave][o][lane]);
    }
    __syncthreads();
  }

  for (int t = tid; t < NN; t += 256) atomicAdd(&P[b * NN + t], P_s[t]);
}

// ---------------------------------------------------------------------------
// Kernel 4: act = squash(P + bias)
// ---------------------------------------------------------------------------
__global__ __launch_bounds__(256) void squash_k(const float* __restrict__ P,
                                                const float* __restrict__ biases,
                                                float* __restrict__ act) {
  const int idx = blockIdx.x * 256 + threadIdx.x;
  const float x  = P[idx] + biases[idx & (NN - 1)];
  const float ss = wave_allsum(x * x);
  act[idx] = x * sqrtf(ss) / (1.0f + ss);
}

// ---------------------------------------------------------------------------
// Kernel 5: out[b,o] = ||squash(P2+bias)|| = ss/(1+ss)
// ---------------------------------------------------------------------------
__global__ __launch_bounds__(256) void finalize_k(const float* __restrict__ P,
                                                  const float* __restrict__ biases,
                                                  float* __restrict__ out) {
  const int idx = blockIdx.x * 256 + threadIdx.x;
  const float x  = P[idx] + biases[idx & (NN - 1)];
  const float ss = wave_allsum(x * x);
  if ((threadIdx.x & 31) == 0) out[idx >> 5] = ss / (1.0f + ss);
}

extern "C" void kernel_launch(void* const* d_in, const int* in_sizes, int n_in,
                              void* d_out, int out_size, void* d_ws, size_t ws_size,
                              hipStream_t stream) {
  const float* inp    = (const float*)d_in[0];  // [64,1024,64]
  const float* W      = (const float*)d_in[1];  // [1024,64,2048]
  const float* biases = (const float*)d_in[2];  // [64,32]
  float* out = (float*)d_out;                   // [64,64]

  char* ws = (char*)d_ws;
  const size_t VOTES_BYTES  = (size_t)BB * II * NN * sizeof(__bf16);  // 268435456
  const size_t LOGITS_BYTES = (size_t)BB * II * OO * sizeof(float);   // 16777216
  const size_t ACT_BYTES    = (size_t)BB * NN * sizeof(float);        // 524288

  __bf16* votes  = (__bf16*)ws;
  float*  logits = (float*)(ws + VOTES_BYTES);
  float*  act0   = (float*)(ws + VOTES_BYTES + LOGITS_BYTES);
  float*  act1   = (float*)(ws + VOTES_BYTES + LOGITS_BYTES + ACT_BYTES);
  float*  P1     = (float*)(ws + VOTES_BYTES + LOGITS_BYTES + 2 * ACT_BYTES);
  float*  P2     = (float*)(ws + VOTES_BYTES + LOGITS_BYTES + 3 * ACT_BYTES);

  // zero the atomic accumulators (P1,P2 contiguous)
  hipMemsetAsync(P1, 0, 2 * ACT_BYTES, stream);

  votes_gemm  <<<II, 256, 0, stream>>>(inp, W, votes);
  reduce_act0 <<<(BB * NN) / 256, 256, 0, stream>>>(votes, biases, act0);
  routing_pass<<<BB * 8, 256, 0, stream>>>(votes, act0, logits, nullptr, P1);
  squash_k    <<<(BB * NN) / 256, 256, 0, stream>>>(P1, biases, act1);
  routing_pass<<<BB * 8, 256, 0, stream>>>(votes, act1, nullptr, logits, P2);
  finalize_k  <<<(BB * NN) / 256, 256, 0, stream>>>(P2, biases, out);
}
